// GCNBlock_32667521253433
// MI455X (gfx1250) — matrix operand order repaired and checked by reading, not compiled, not run on hardware
//
#include <hip/hip_runtime.h>
#include <hip/hip_bf16.h>
#include <math.h>

typedef __attribute__((ext_vector_type(16))) _Float16 v16h;
typedef __attribute__((ext_vector_type(8)))  float    v8f;

#define FEAT 128
#define LN_EPS 1e-5f

// ---------------- degree / normalization ----------------
__global__ void zero_f32_kernel(float* __restrict__ p, int n) {
  int i = blockIdx.x * blockDim.x + threadIdx.x;
  if (i < n) p[i] = 0.0f;
}

__global__ void deg_accum_kernel(const long long* __restrict__ dst,
                                 float* __restrict__ deg, int E) {
  int i = blockIdx.x * blockDim.x + threadIdx.x;
  if (i < E) {
    int d = (int)dst[i];
    __hip_atomic_fetch_add(&deg[d], 1.0f, __ATOMIC_RELAXED, __HIP_MEMORY_SCOPE_AGENT);
  }
}

// deg -> deg^{-1/2}, including the +1 self-loop (so deg >= 1 always)
__global__ void deg_to_dinv_kernel(float* __restrict__ deg, int N) {
  int i = blockIdx.x * blockDim.x + threadIdx.x;
  if (i < N) deg[i] = rsqrtf(deg[i] + 1.0f);
}

// ---------------- H = X(Nrows x 128) @ W(128 x 128), f16 WMMA, f32 accum ----
// One block = 256 threads = 8 wave32; block handles a 16-row tile of X;
// wave w produces the 16x16 output tile at columns [16w, 16w+16).
__global__ __launch_bounds__(256) void gemm128_wmma_kernel(
    const float* __restrict__ X, const float* __restrict__ W,
    float* __restrict__ H, int Nrows)
{
  __shared__ alignas(16) _Float16 Ws[128][136];   // +8 halves pad (keeps 16B align)
  __shared__ alignas(16) _Float16 Xs[16][136];

  const int tid = threadIdx.x;
  const int m0  = blockIdx.x * 16;

  // stage W (128x128) into LDS as f16, float4 granularity (4096 vec4 loads)
  {
    const float4* W4 = (const float4*)W;
    for (int idx = tid; idx < 128 * 32; idx += 256) {
      float4 w = W4[idx];
      int r  = idx >> 5;
      int c  = (idx & 31) << 2;
      Ws[r][c + 0] = (_Float16)w.x;
      Ws[r][c + 1] = (_Float16)w.y;
      Ws[r][c + 2] = (_Float16)w.z;
      Ws[r][c + 3] = (_Float16)w.w;
    }
  }
  // stage X tile (16x128) into LDS as f16 (512 vec4 loads; full-tile fast path)
  if (m0 + 16 <= Nrows) {
    const float4* X4 = (const float4*)(X + (size_t)m0 * FEAT);
    for (int idx = tid; idx < 16 * 32; idx += 256) {
      float4 v = X4[idx];
      int r = idx >> 5;
      int c = (idx & 31) << 2;
      Xs[r][c + 0] = (_Float16)v.x;
      Xs[r][c + 1] = (_Float16)v.y;
      Xs[r][c + 2] = (_Float16)v.z;
      Xs[r][c + 3] = (_Float16)v.w;
    }
  } else {
    for (int idx = tid; idx < 16 * 128; idx += 256) {
      int r = idx >> 7, c = idx & 127;
      int row = m0 + r;
      Xs[r][c] = (row < Nrows) ? (_Float16)X[(size_t)row * FEAT + c]
                               : (_Float16)0.0f;
    }
  }
  __syncthreads();

  const int wv   = tid >> 5;        // 0..7 : output N-tile
  const int lane = tid & 31;
  const int mrow = lane & 15;       // A: M row, B/C: N column held by this lane
  const int khi  = (lane >> 4) * 8; // A: K sub-block select per lane half

  v8f acc = {};
  #pragma unroll
  for (int t = 0; t < 4; ++t) {     // K = 128 in 4 steps of 32
    v16h a, b;
    // A fragment (16x32 f16): lane m, halves 2r+s -> K = 32t + 16*(r/4) + khi + 2*(r%4) + s
    #pragma unroll
    for (int r = 0; r < 8; ++r) {
      int ka = 32 * t + ((r >> 2) << 4) + khi + ((r & 3) << 1);
      a[2 * r]     = Xs[mrow][ka];
      a[2 * r + 1] = Xs[mrow][ka + 1];
    }
    // B fragment (32x16 f16): lane n, halves 2r+s -> K = 32t + 16*(r/4) + khi + 2*(r%4) + s
    #pragma unroll
    for (int r = 0; r < 8; ++r) {
      int kb = 32 * t + ((r >> 2) << 4) + khi + ((r & 3) << 1);
      b[2 * r]     = Ws[kb][wv * 16 + mrow];
      b[2 * r + 1] = Ws[kb + 1][wv * 16 + mrow];
    }
    acc = __builtin_amdgcn_wmma_f32_16x16x32_f16(
        /*neg_a=*/false, a, /*neg_b=*/false, b,
        /*c_mod=*/(short)0, acc, /*reuse_a=*/false, /*reuse_b=*/false);
  }

  // C/D layout: VGPR g -> M = g + 8*(lane>=16), N = lane & 15
  float* Hout = H + (size_t)m0 * FEAT + wv * 16 + (lane & 15)
                + (size_t)((lane >> 4) << 3) * FEAT;
  if (m0 + 16 <= Nrows) {
    #pragma unroll
    for (int g = 0; g < 8; ++g)
      Hout[(size_t)g * FEAT] = acc[g];
  } else {
    #pragma unroll
    for (int g = 0; g < 8; ++g) {
      int row = m0 + g + ((lane >> 4) << 3);
      if (row < Nrows) Hout[(size_t)g * FEAT] = acc[g];
    }
  }
}

// ---------------- out = h * dinv^2 (self loop) + bias ; fully inits out -----
__global__ void init_out_kernel(const float* __restrict__ h,
                                const float* __restrict__ dinv,
                                const float* __restrict__ bias,
                                float* __restrict__ out, int N) {
  int idx = blockIdx.x * blockDim.x + threadIdx.x;
  if (idx < N * FEAT) {
    int node = idx >> 7;
    int f    = idx & 127;
    float di = dinv[node];
    out[idx] = h[idx] * di * di + bias[f];
  }
}

// ---------------- per-edge scatter: one wave per edge, float4 lanes ---------
__global__ __launch_bounds__(256) void edge_scatter_kernel(
    const long long* __restrict__ src, const long long* __restrict__ dst,
    const float* __restrict__ dinv, const float* __restrict__ h,
    float* __restrict__ out, int E)
{
  int e = blockIdx.x * 8 + (threadIdx.x >> 5);
  if (e >= E) return;
  int lane = threadIdx.x & 31;
  int s = (int)src[e];
  int d = (int)dst[e];
  float w = dinv[s] * dinv[d];

  const float4 v = ((const float4*)(h + (size_t)s * FEAT))[lane];
  float* o = out + (size_t)d * FEAT + lane * 4;
  __hip_atomic_fetch_add(o + 0, v.x * w, __ATOMIC_RELAXED, __HIP_MEMORY_SCOPE_AGENT);
  __hip_atomic_fetch_add(o + 1, v.y * w, __ATOMIC_RELAXED, __HIP_MEMORY_SCOPE_AGENT);
  __hip_atomic_fetch_add(o + 2, v.z * w, __ATOMIC_RELAXED, __HIP_MEMORY_SCOPE_AGENT);
  __hip_atomic_fetch_add(o + 3, v.w * w, __ATOMIC_RELAXED, __HIP_MEMORY_SCOPE_AGENT);
}

// ---------------- fused LayerNorm + ELU, one wave per 128-feature row -------
__global__ __launch_bounds__(256) void ln_elu_kernel(
    float* __restrict__ io, const float* __restrict__ gamma,
    const float* __restrict__ beta, int N)
{
  int row = blockIdx.x * 8 + (threadIdx.x >> 5);
  if (row >= N) return;
  int lane = threadIdx.x & 31;

  float4 v = ((const float4*)(io + (size_t)row * FEAT))[lane];
  float s  = v.x + v.y + v.z + v.w;
  float sq = v.x * v.x + v.y * v.y + v.z * v.z + v.w * v.w;
  #pragma unroll
  for (int off = 16; off > 0; off >>= 1) {
    s  += __shfl_xor(s,  off, 32);
    sq += __shfl_xor(sq, off, 32);
  }
  float mu  = s * (1.0f / FEAT);
  float var = sq * (1.0f / FEAT) - mu * mu;
  float inv = rsqrtf(var + LN_EPS);

  float4 g = ((const float4*)gamma)[lane];
  float4 b = ((const float4*)beta)[lane];
  float4 r;
  r.x = (v.x - mu) * inv * g.x + b.x;
  r.y = (v.y - mu) * inv * g.y + b.y;
  r.z = (v.z - mu) * inv * g.z + b.z;
  r.w = (v.w - mu) * inv * g.w + b.w;
  // ELU (alpha = 1)
  r.x = r.x > 0.0f ? r.x : expm1f(r.x);
  r.y = r.y > 0.0f ? r.y : expm1f(r.y);
  r.z = r.z > 0.0f ? r.z : expm1f(r.z);
  r.w = r.w > 0.0f ? r.w : expm1f(r.w);
  ((float4*)(io + (size_t)row * FEAT))[lane] = r;
}

// ---------------- host-side launch sequence --------------------------------
extern "C" void kernel_launch(void* const* d_in, const int* in_sizes, int n_in,
                              void* d_out, int out_size, void* d_ws, size_t ws_size,
                              hipStream_t stream) {
  const float*     x    = (const float*)d_in[0];
  const long long* ei   = (const long long*)d_in[1];  // int64 (2, E)
  const float*     W1   = (const float*)d_in[2];
  const float*     b1   = (const float*)d_in[3];
  const float*     g1   = (const float*)d_in[4];
  const float*     be1  = (const float*)d_in[5];
  const float*     W2   = (const float*)d_in[6];
  const float*     b2   = (const float*)d_in[7];
  const float*     g2   = (const float*)d_in[8];
  const float*     be2  = (const float*)d_in[9];

  const int N = in_sizes[0] / FEAT;      // 50000
  const int E = in_sizes[1] / 2;         // 1.6M
  const long long* src = ei;
  const long long* dst = ei + E;

  float* out  = (float*)d_out;
  float* wsH  = (float*)d_ws;                    // N x 128 GEMM output
  float* dinv = wsH + (size_t)N * FEAT;          // N floats (deg -> dinv)

  const int T = 256;
  const int gN     = (N + T - 1) / T;
  const int gE     = (E + T - 1) / T;
  const int gNF    = (N * FEAT + T - 1) / T;
  const int gGemm  = (N + 15) / 16;
  const int gWaveE = (E + 7) / 8;                // 8 waves / block
  const int gWaveN = (N + 7) / 8;

  // degrees -> dinv
  zero_f32_kernel<<<gN, T, 0, stream>>>(dinv, N);
  deg_accum_kernel<<<gE, T, 0, stream>>>(dst, dinv, E);
  deg_to_dinv_kernel<<<gN, T, 0, stream>>>(dinv, N);

  // ---- layer 1 ----
  gemm128_wmma_kernel<<<gGemm, T, 0, stream>>>(x, W1, wsH, N);
  init_out_kernel<<<gNF, T, 0, stream>>>(wsH, dinv, b1, out, N);
  edge_scatter_kernel<<<gWaveE, T, 0, stream>>>(src, dst, dinv, wsH, out, E);
  ln_elu_kernel<<<gWaveN, T, 0, stream>>>(out, g1, be1, N);

  // ---- layer 2 (input is d_out from layer 1) ----
  gemm128_wmma_kernel<<<gGemm, T, 0, stream>>>(out, W2, wsH, N);
  init_out_kernel<<<gNF, T, 0, stream>>>(wsH, dinv, b2, out, N);
  edge_scatter_kernel<<<gWaveE, T, 0, stream>>>(src, dst, dinv, wsH, out, E);
  ln_elu_kernel<<<gWaveN, T, 0, stream>>>(out, g2, be2, N);
}
